// GAT_44624710205913
// MI455X (gfx1250) — matrix operand order repaired and checked by reading, not compiled, not run on hardware
//
#include <hip/hip_runtime.h>
#include <hip/hip_bf16.h>
#include <string.h>

typedef __attribute__((ext_vector_type(16))) _Float16 v16h;
typedef __attribute__((ext_vector_type(8)))  _Float16 v8h;
typedef __attribute__((ext_vector_type(8)))  float    v8f;

// ---------------- problem constants (from reference) ----------------
#define BGR   256
#define NPER  400
#define DEG   16
#define INC   200
#define KP1   200
#define KP2   100
static const int  NN  = BGR * NPER;       // 102400 nodes
static const long EE  = (long)NN * DEG;   // 1638400 edges
static const int  NN2 = BGR * KP1;        // 51200 nodes after pool1
static const int  NN3 = BGR * KP2;        // 25600 nodes after pool2

// ---------------- helpers ----------------
__device__ __forceinline__ unsigned f2ord(float f) {
    unsigned u = __float_as_uint(f);
    return (u & 0x80000000u) ? ~u : (u | 0x80000000u);
}
__device__ __forceinline__ float ord2f(unsigned u) {
    u = (u & 0x80000000u) ? (u & 0x7fffffffu) : ~u;
    return __uint_as_float(u);
}
static inline unsigned h_f2ord(float f) {
    unsigned u; memcpy(&u, &f, 4);
    return (u & 0x80000000u) ? ~u : (u | 0x80000000u);
}
static inline int roundup32(int k) { return (k + 31) & ~31; }

// ---------------- GEMM operand staging ----------------
// A[M,K] f32 -> Af[M,Kp] f16, zero-padded K (Kp multiple of 32; M multiple of 16)
__global__ void k_a2h(const float* __restrict__ A, _Float16* __restrict__ Af,
                      long M, int K, int Kp)
{
    long i = blockIdx.x * (long)blockDim.x + threadIdx.x;
    if (i >= M * Kp) return;
    int  k = (int)(i % Kp);
    long m = i / Kp;
    Af[i] = (_Float16)(k < K ? A[m * K + k] : 0.f);
}

// W[K,N] f32 -> Bf in per-lane B-fragment order (contiguous k, as A uses): [ntile][kchunk][lane][i]
// lane half (lane>>4) holds K = kc*32 + half*16 + i; column = ntile*16 + (lane&15)
__global__ void k_w2frag(const float* __restrict__ W, _Float16* __restrict__ Bf,
                         int K, int N, int Kp)
{
    int  Kc  = Kp / 32;
    int  ntp = (N + 15) / 16;
    long total = (long)ntp * Kc * 512;
    long idx = blockIdx.x * (long)blockDim.x + threadIdx.x;
    if (idx >= total) return;
    int  i    = (int)(idx & 15);
    int  lane = (int)((idx >> 4) & 31);
    long r    = idx >> 9;                 // nt*Kc + kc
    int  nt   = (int)(r / Kc), kc = (int)(r % Kc);
    int  k    = kc * 32 + (lane >> 4) * 16 + i;
    int  n    = nt * 16 + (lane & 15);
    Bf[idx] = (_Float16)((k < K && n < N) ? W[(long)k * N + n] : 0.f);
}

// ---------------- WMMA GEMM: C = act(Af[M,Kp] @ W + bias), optional eval-BN ----------------
// One wave -> one 16x16 tile. B panel for this block's ntile staged in LDS once.
// Requires: M % 16 == 0, Kp % 32 == 0, Kp <= 512.
__global__ void __launch_bounds__(256)
k_wmma_gemm(const _Float16* __restrict__ Af, const _Float16* __restrict__ Bf,
            const float* __restrict__ bias, const float* __restrict__ bng,
            const float* __restrict__ bnb, float bninv,
            float* __restrict__ C, int M, int N, int Kp, int act)
{
    __shared__ __attribute__((aligned(32))) _Float16 bsh[8192];   // up to Kc=16 chunks (16 KB)
    const int Kc = Kp >> 5;

    // cooperative copy of this ntile's B panel (Kc*512 f16) into LDS
    {
        const v8h* bs = (const v8h*)(Bf + (size_t)blockIdx.y * Kc * 512);
        v8h*       bd = (v8h*)bsh;
        int cnt = Kc * 64;                                        // v8h units
        for (int i = threadIdx.x; i < cnt; i += 256) bd[i] = bs[i];
    }
    __syncthreads();

    const int lane  = threadIdx.x & 31;
    const int wave  = threadIdx.x >> 5;
    const int m0    = (blockIdx.x * 8 + wave) * 16;
    if (m0 >= M) return;                      // wave-uniform; EXEC all-1s for WMMA
    const int n0   = blockIdx.y * 16;
    const int half = lane >> 4;
    const int l16  = lane & 15;

    const _Float16* ap = Af + (size_t)(m0 + l16) * Kp + half * 16;
    v8f acc = {};
    for (int kc = 0; kc < Kc; ++kc) {
        const int k0 = kc * 32;
        v8h alo = *(const v8h*)(ap + k0);
        v8h ahi = *(const v8h*)(ap + k0 + 8);
        __builtin_prefetch(ap + k0 + 64, 0, 1);                   // global_prefetch_b8
        v16h a = __builtin_shufflevector(alo, ahi,
                 0,1,2,3,4,5,6,7,8,9,10,11,12,13,14,15);
        v16h b = *(const v16h*)(bsh + ((kc << 5) + lane) * 16);   // ds_load
        acc = __builtin_amdgcn_wmma_f32_16x16x32_f16(false, a, false, b,
                                                     (short)0, acc, false, false);
    }

    const int col = n0 + l16;
    if (col < N) {
        float bv = bias ? bias[col] : 0.f;
        float g  = bng  ? bng[col]  : 1.f;
        float bb = bnb  ? bnb[col]  : 0.f;
#pragma unroll
        for (int d = 0; d < 8; ++d) {
            int row = m0 + d + 8 * half;                          // always < M (M%16==0)
            float v = acc[d] + bv;
            if (act == 1) v = fmaxf(v, 0.f);
            if (bng) v = g * (v * bninv) + bb;
            C[(long)row * N + col] = v;
        }
    }
}

// ---------------- attention projections: e_s/e_d per (node,head) ----------------
__global__ void k_att_proj(const float* __restrict__ xp, const float* __restrict__ asrc,
                           const float* __restrict__ adst, float* __restrict__ es,
                           float* __restrict__ ed, int Nn, int H, int Cc)
{
    long i = blockIdx.x * (long)blockDim.x + threadIdx.x;   // n*H + h
    if (i >= (long)Nn * H) return;
    int h = (int)(i % H);
    const float* row = xp + i * Cc;
    float ss = 0.f, dd = 0.f;
    for (int c = 0; c < Cc; ++c) {
        float v = row[c];
        ss += v * asrc[h * Cc + c];
        dd += v * adst[h * Cc + c];
    }
    es[i] = ss; ed[i] = dd;
}

// ---------------- edge softmax passes (edges [0,E) + self-loops [E,E+Nn)) ----------------
__global__ void k_edge_max(const int* __restrict__ src, const int* __restrict__ dst,
                           const int* __restrict__ msk, const float* __restrict__ es,
                           const float* __restrict__ ed, unsigned* __restrict__ smax,
                           long E, int Nn, int H)
{
    long i = blockIdx.x * (long)blockDim.x + threadIdx.x;
    if (i >= E + Nn) return;
    int s, d, m;
    if (i < E) { s = src[i]; d = dst[i]; m = msk ? msk[i] : 1; }
    else       { s = d = (int)(i - E);   m = 1; }
    if (!m) return;
    for (int h = 0; h < H; ++h) {
        float l = es[(long)s * H + h] + ed[(long)d * H + h];
        l = l > 0.f ? l : 0.2f * l;                       // leaky_relu(0.2)
        atomicMax(&smax[(long)d * H + h], f2ord(l));
    }
}

__global__ void k_edge_sum(const int* __restrict__ src, const int* __restrict__ dst,
                           const int* __restrict__ msk, const float* __restrict__ es,
                           const float* __restrict__ ed, const unsigned* __restrict__ smax,
                           float* __restrict__ den, long E, int Nn, int H)
{
    long i = blockIdx.x * (long)blockDim.x + threadIdx.x;
    if (i >= E + Nn) return;
    int s, d, m;
    if (i < E) { s = src[i]; d = dst[i]; m = msk ? msk[i] : 1; }
    else       { s = d = (int)(i - E);   m = 1; }
    if (!m) return;
    for (int h = 0; h < H; ++h) {
        float l = es[(long)s * H + h] + ed[(long)d * H + h];
        l = l > 0.f ? l : 0.2f * l;
        atomicAdd(&den[(long)d * H + h], expf(l - ord2f(smax[(long)d * H + h])));
    }
}

__global__ void k_edge_agg(const int* __restrict__ src, const int* __restrict__ dst,
                           const int* __restrict__ msk, const float* __restrict__ es,
                           const float* __restrict__ ed, const unsigned* __restrict__ smax,
                           const float* __restrict__ den, const float* __restrict__ xp,
                           float* __restrict__ agg, long E, int Nn, int H, int Cc)
{
    long i = blockIdx.x * (long)blockDim.x + threadIdx.x;   // (edge)*H + h
    if (i >= (E + Nn) * H) return;
    long e = i / H; int h = (int)(i % H);
    int s, d, m;
    if (e < E) { s = src[e]; d = dst[e]; m = msk ? msk[e] : 1; }
    else       { s = d = (int)(e - E);   m = 1; }
    if (!m) return;
    float l = es[(long)s * H + h] + ed[(long)d * H + h];
    l = l > 0.f ? l : 0.2f * l;
    float w  = expf(l - ord2f(smax[(long)d * H + h]));
    float dn = den[(long)d * H + h];
    float alpha = w / (dn > 1e-16f ? dn : 1e-16f);
    const float* xs = xp + ((long)s * H + h) * Cc;
    float*       od = agg + ((long)d * H + h) * Cc;
    for (int c = 0; c < Cc; ++c) atomicAdd(&od[c], alpha * xs[c]);
}

// ---------------- finalize: h = act(agg + bias) in place (act: 0 none, 2 elu) ----------------
__global__ void k_bias_act(float* __restrict__ h, const float* __restrict__ bias,
                           int Nn, int Cfull, int act)
{
    long i = blockIdx.x * (long)blockDim.x + threadIdx.x;
    if (i >= (long)Nn * Cfull) return;
    int c = (int)(i % Cfull);
    float v = h[i] + bias[c];
    if (act == 2) v = v > 0.f ? v : (expf(v) - 1.f);      // elu
    h[i] = v;
}

// ---------------- topk pooling pieces ----------------
__global__ void k_score(const float* __restrict__ h, const float* __restrict__ w,
                        float* __restrict__ score, int Nn, int Cc)
{
    long n = blockIdx.x * (long)blockDim.x + threadIdx.x;
    if (n >= Nn) return;
    float nw = 0.f, dot = 0.f;
    for (int c = 0; c < Cc; ++c) { float wc = w[c]; nw += wc * wc; dot += h[n * (long)Cc + c] * wc; }
    score[n] = tanhf(dot * rsqrtf(nw));
}

// block per graph; exact jax.lax.top_k semantics (descending, stable)
__global__ void k_topk(const float* __restrict__ score, int n_per, int k,
                       int* __restrict__ perm, float* __restrict__ vals)
{
    __shared__ float sc[NPER];
    int g = blockIdx.x;
    for (int i = threadIdx.x; i < n_per; i += blockDim.x)
        sc[i] = score[(long)g * n_per + i];
    __syncthreads();
    for (int i = threadIdx.x; i < n_per; i += blockDim.x) {
        float si = sc[i];
        int rank = 0;
        for (int j = 0; j < n_per; ++j) {
            float sj = sc[j];
            rank += (sj > si) || (sj == si && j < i);
        }
        if (rank < k) {
            perm[(long)g * k + rank] = g * n_per + i;   // global node index
            vals[(long)g * k + rank] = si;
        }
    }
}

__global__ void k_fill_u32(unsigned* __restrict__ p, unsigned v, long n)
{
    long i = blockIdx.x * (long)blockDim.x + threadIdx.x;
    if (i < n) p[i] = v;
}

__global__ void k_scatter_map(const int* __restrict__ perm, int* __restrict__ map, int nk)
{
    int j = blockIdx.x * blockDim.x + threadIdx.x;
    if (j < nk) map[perm[j]] = j;
}

__global__ void k_pool_apply(const float* __restrict__ h, const int* __restrict__ perm,
                             const float* __restrict__ vals, float* __restrict__ xk,
                             int nk, int Cc)
{
    long i = blockIdx.x * (long)blockDim.x + threadIdx.x;
    if (i >= (long)nk * Cc) return;
    int j = (int)(i / Cc), c = (int)(i % Cc);
    xk[i] = h[(long)perm[j] * Cc + c] * vals[j];
}

__global__ void k_relabel(const int* __restrict__ src, const int* __restrict__ dst,
                          const int* __restrict__ msk_in, const int* __restrict__ map,
                          int* __restrict__ s2, int* __restrict__ d2, int* __restrict__ m2,
                          long E)
{
    long e = blockIdx.x * (long)blockDim.x + threadIdx.x;
    if (e >= E) return;
    int m = msk_in ? msk_in[e] : 1;
    int a = map[src[e]], b = map[dst[e]];
    m = m && a >= 0 && b >= 0;
    s2[e] = m ? a : 0; d2[e] = m ? b : 0; m2[e] = m;
}

// ---------------- readout: [max(k), mean(k)] per graph, 64 ch each ----------------
__global__ void k_readout(const float* __restrict__ h, float* __restrict__ xg,
                          int nb, int k, int addf)
{
    long i = blockIdx.x * (long)blockDim.x + threadIdx.x;   // b*64 + c
    if (i >= (long)nb * 64) return;
    int b = (int)(i / 64), c = (int)(i % 64);
    float mx = -3.4e38f, sm = 0.f;
    const float* p = h + ((long)b * k) * 64 + c;
    for (int j = 0; j < k; ++j) { float v = p[(long)j * 64]; mx = fmaxf(mx, v); sm += v; }
    float mean = sm / (float)k;
    float* o = xg + (long)b * 128;
    if (addf) { o[c] += mx; o[64 + c] += mean; }
    else      { o[c]  = mx; o[64 + c]  = mean; }
}

// ---------------- tails ----------------
__global__ void k_lsm2(const float* __restrict__ in, float* __restrict__ out, int nb)
{
    int b = blockIdx.x * blockDim.x + threadIdx.x;
    if (b >= nb) return;
    float x0 = in[b * 2], x1 = in[b * 2 + 1];
    float m = fmaxf(x0, x1);
    float l = logf(expf(x0 - m) + expf(x1 - m)) + m;
    out[b * 2] = x0 - l; out[b * 2 + 1] = x1 - l;
}

__global__ void k_sigmoid(const float* __restrict__ in, float* __restrict__ out, long n)
{
    long i = blockIdx.x * (long)blockDim.x + threadIdx.x;
    if (i < n) out[i] = 1.f / (1.f + expf(-in[i]));
}

// =====================================================================
extern "C" void kernel_launch(void* const* d_in, const int* in_sizes, int n_in,
                              void* d_out, int out_size, void* d_ws, size_t ws_size,
                              hipStream_t stream)
{
    (void)in_sizes; (void)n_in; (void)out_size; (void)ws_size;
    const float* x       = (const float*)d_in[0];
    const int*   ei      = (const int*)  d_in[1];
    const int*   src     = ei;
    const int*   dst     = ei + EE;
    const float* W1      = (const float*)d_in[2];
    const float* a1_src  = (const float*)d_in[3];
    const float* a1_dst  = (const float*)d_in[4];
    const float* b1      = (const float*)d_in[5];
    const float* W2      = (const float*)d_in[6];
    const float* a2_src  = (const float*)d_in[7];
    const float* a2_dst  = (const float*)d_in[8];
    const float* b2      = (const float*)d_in[9];
    const float* pool1_w = (const float*)d_in[10];
    const float* pool2_w = (const float*)d_in[11];
    const float* fc1_w   = (const float*)d_in[12];
    const float* fc1_b   = (const float*)d_in[13];
    const float* bn1_g   = (const float*)d_in[14];
    const float* bn1_b   = (const float*)d_in[15];
    const float* fc2_w   = (const float*)d_in[16];
    const float* fc2_b   = (const float*)d_in[17];
    const float* bn2_g   = (const float*)d_in[18];
    const float* bn2_b   = (const float*)d_in[19];
    const float* fc3_w   = (const float*)d_in[20];
    const float* fc3_b   = (const float*)d_in[21];
    float* out = (float*)d_out;

    // bump allocator over workspace
    char* wsb = (char*)d_ws;
    size_t off = 0;
    auto alloc = [&](size_t bytes) -> void* {
        void* p = wsb + off;
        off = (off + bytes + 255) & ~(size_t)255;
        return p;
    };
    const int KpIN = roundup32(INC);   // 224
    float*    xp1   = (float*)   alloc((size_t)NN * 64 * 4);
    float*    es1   = (float*)   alloc((size_t)NN * 8 * 4);
    float*    ed1   = (float*)   alloc((size_t)NN * 8 * 4);
    unsigned* smax1 = (unsigned*)alloc((size_t)NN * 8 * 4);
    float*    den1  = (float*)   alloc((size_t)NN * 8 * 4);
    float*    h1    = (float*)   alloc((size_t)NN * 64 * 4);   // agg then in-place elu(+b1)
    float*    sc1   = (float*)   alloc((size_t)NN * 4);
    int*      perm1 = (int*)     alloc((size_t)NN2 * 4);
    float*    vals1 = (float*)   alloc((size_t)NN2 * 4);
    int*      map1  = (int*)     alloc((size_t)NN * 4);
    int*      src2  = (int*)     alloc((size_t)EE * 4);
    int*      dst2  = (int*)     alloc((size_t)EE * 4);
    int*      msk2  = (int*)     alloc((size_t)EE * 4);
    float*    xk1   = (float*)   alloc((size_t)NN2 * 64 * 4);
    float*    xg    = (float*)   alloc((size_t)BGR * 128 * 4);
    float*    xp2   = (float*)   alloc((size_t)NN2 * 64 * 4);
    float*    es2   = (float*)   alloc((size_t)NN2 * 4);
    float*    ed2   = (float*)   alloc((size_t)NN2 * 4);
    unsigned* smax2 = (unsigned*)alloc((size_t)NN2 * 4);
    float*    den2  = (float*)   alloc((size_t)NN2 * 4);
    float*    h2    = (float*)   alloc((size_t)NN2 * 64 * 4);
    float*    sc2   = (float*)   alloc((size_t)NN2 * 4);
    int*      perm2 = (int*)     alloc((size_t)NN3 * 4);
    float*    vals2 = (float*)   alloc((size_t)NN3 * 4);
    float*    xk2   = (float*)   alloc((size_t)NN3 * 64 * 4);
    float*    f1o   = (float*)   alloc((size_t)BGR * 256 * 4);
    float*    f2o   = (float*)   alloc((size_t)BGR * 512 * 4);
    float*    f3o   = (float*)   alloc((size_t)BGR * 2 * 4);
    // f16 staging for WMMA GEMMs
    _Float16* Afx   = (_Float16*)alloc((size_t)NN  * KpIN * 2);   // x padded
    _Float16* Afk1  = (_Float16*)alloc((size_t)NN2 * 64  * 2);    // xk1
    _Float16* Afxg  = (_Float16*)alloc((size_t)BGR * 128 * 2);    // xg
    _Float16* Aff1  = (_Float16*)alloc((size_t)BGR * 256 * 2);    // f1o
    _Float16* Aff2  = (_Float16*)alloc((size_t)BGR * 512 * 2);    // f2o
    _Float16* BfW1  = (_Float16*)alloc((size_t)4  * (KpIN/32) * 512 * 2);
    _Float16* BfW2  = (_Float16*)alloc((size_t)4  * 2  * 512 * 2);
    _Float16* Bff1  = (_Float16*)alloc((size_t)16 * 4  * 512 * 2);
    _Float16* Bff2  = (_Float16*)alloc((size_t)32 * 8  * 512 * 2);
    _Float16* Bff3  = (_Float16*)alloc((size_t)1  * 16 * 512 * 2);

    const float bninv  = 1.0f / sqrtf(1.0f + 1e-5f);
    const unsigned NEG = h_f2ord(-1e9f);
    auto blocks = [](long n) { return dim3((unsigned)((n + 255) / 256)); };
    auto gemm = [&](const float* A, _Float16* Af, const float* W, _Float16* Bf,
                    const float* bias, const float* bg, const float* bb,
                    float* C, int M, int N, int K, int act) {
        int Kp  = roundup32(K);
        int ntp = (N + 15) / 16;
        k_a2h<<<blocks((long)M * Kp), 256, 0, stream>>>(A, Af, M, K, Kp);
        k_w2frag<<<blocks((long)ntp * (Kp / 32) * 512), 256, 0, stream>>>(W, Bf, K, N, Kp);
        k_wmma_gemm<<<dim3((M / 16 + 7) / 8, ntp), 256, 0, stream>>>(
            Af, Bf, bias, bg, bb, bninv, C, M, N, Kp, act);
    };

    // ---- GAT layer 1 ----
    gemm(x, Afx, W1, BfW1, nullptr, nullptr, nullptr, xp1, NN, 64, INC, 0);
    k_att_proj<<<blocks((long)NN * 8), 256, 0, stream>>>(xp1, a1_src, a1_dst, es1, ed1, NN, 8, 8);
    k_fill_u32<<<blocks((long)NN * 8), 256, 0, stream>>>(smax1, NEG, (long)NN * 8);
    k_fill_u32<<<blocks((long)NN * 8), 256, 0, stream>>>((unsigned*)den1, 0u, (long)NN * 8);
    k_fill_u32<<<blocks((long)NN * 64), 256, 0, stream>>>((unsigned*)h1, 0u, (long)NN * 64);
    k_edge_max<<<blocks(EE + NN), 256, 0, stream>>>(src, dst, nullptr, es1, ed1, smax1, EE, NN, 8);
    k_edge_sum<<<blocks(EE + NN), 256, 0, stream>>>(src, dst, nullptr, es1, ed1, smax1, den1, EE, NN, 8);
    k_edge_agg<<<blocks((EE + NN) * 8), 256, 0, stream>>>(src, dst, nullptr, es1, ed1, smax1,
                                                          den1, xp1, h1, EE, NN, 8, 8);
    k_bias_act<<<blocks((long)NN * 64), 256, 0, stream>>>(h1, b1, NN, 64, 2 /*elu*/);

    // ---- pool 1 + readout 1 ----
    k_score<<<blocks(NN), 256, 0, stream>>>(h1, pool1_w, sc1, NN, 64);
    k_topk<<<BGR, 256, 0, stream>>>(sc1, NPER, KP1, perm1, vals1);
    k_fill_u32<<<blocks(NN), 256, 0, stream>>>((unsigned*)map1, 0xFFFFFFFFu, NN);
    k_scatter_map<<<blocks(NN2), 256, 0, stream>>>(perm1, map1, NN2);
    k_pool_apply<<<blocks((long)NN2 * 64), 256, 0, stream>>>(h1, perm1, vals1, xk1, NN2, 64);
    k_relabel<<<blocks(EE), 256, 0, stream>>>(src, dst, nullptr, map1, src2, dst2, msk2, EE);
    k_readout<<<blocks((long)BGR * 64), 256, 0, stream>>>(xk1, xg, BGR, KP1, 0);

    // ---- GAT layer 2 (H=1, C=64) ----
    gemm(xk1, Afk1, W2, BfW2, nullptr, nullptr, nullptr, xp2, NN2, 64, 64, 0);
    k_att_proj<<<blocks(NN2), 256, 0, stream>>>(xp2, a2_src, a2_dst, es2, ed2, NN2, 1, 64);
    k_fill_u32<<<blocks(NN2), 256, 0, stream>>>(smax2, NEG, NN2);
    k_fill_u32<<<blocks(NN2), 256, 0, stream>>>((unsigned*)den2, 0u, NN2);
    k_fill_u32<<<blocks((long)NN2 * 64), 256, 0, stream>>>((unsigned*)h2, 0u, (long)NN2 * 64);
    k_edge_max<<<blocks(EE + NN2), 256, 0, stream>>>(src2, dst2, msk2, es2, ed2, smax2, EE, NN2, 1);
    k_edge_sum<<<blocks(EE + NN2), 256, 0, stream>>>(src2, dst2, msk2, es2, ed2, smax2, den2, EE, NN2, 1);
    k_edge_agg<<<blocks(EE + NN2), 256, 0, stream>>>(src2, dst2, msk2, es2, ed2, smax2,
                                                     den2, xp2, h2, EE, NN2, 1, 64);
    k_bias_act<<<blocks((long)NN2 * 64), 256, 0, stream>>>(h2, b2, NN2, 64, 0 /*none*/);

    // ---- pool 2 + readout 2 ----
    k_score<<<blocks(NN2), 256, 0, stream>>>(h2, pool2_w, sc2, NN2, 64);
    k_topk<<<BGR, 256, 0, stream>>>(sc2, KP1, KP2, perm2, vals2);
    k_pool_apply<<<blocks((long)NN3 * 64), 256, 0, stream>>>(h2, perm2, vals2, xk2, NN3, 64);
    k_readout<<<blocks((long)BGR * 64), 256, 0, stream>>>(xk2, xg, BGR, KP2, 1 /*add*/);

    // ---- MLP head (relu + eval-BN folded into GEMM epilogue) ----
    gemm(xg,  Afxg, fc1_w, Bff1, fc1_b, bn1_g, bn1_b, f1o, BGR, 256, 128, 1);
    gemm(f1o, Aff1, fc2_w, Bff2, fc2_b, bn2_g, bn2_b, f2o, BGR, 512, 256, 1);
    gemm(f2o, Aff2, fc3_w, Bff3, fc3_b, nullptr, nullptr, f3o, BGR, 2, 512, 0);

    // ---- outputs: log_softmax [B,2], sigmoid(s1) [B,200], sigmoid(s2) [B,100] ----
    k_lsm2<<<blocks(BGR), 256, 0, stream>>>(f3o, out, BGR);
    k_sigmoid<<<blocks(NN2), 256, 0, stream>>>(vals1, out + BGR * 2, NN2);
    k_sigmoid<<<blocks(NN3), 256, 0, stream>>>(vals2, out + BGR * 2 + NN2, NN3);
}
